// MoETransformerClassifier_88665304859103
// MI455X (gfx1250) — matrix operand order repaired and checked by reading, not compiled, not run on hardware
//
#include <hip/hip_runtime.h>

// ---------------------------------------------------------------------------
// MoE + 2-layer transformer classifier for MI455X (gfx1250, wave32, WMMA).
// All GEMMs use V_WMMA_F32_16X16X32_BF16 (bf16 in, f32 accumulate).
// ---------------------------------------------------------------------------

typedef __bf16 bf16;
typedef __attribute__((ext_vector_type(16))) __bf16 v16bf;
typedef __attribute__((ext_vector_type(8)))  float  v8f;

#define LN_EPSF 1e-5f
// Wave-local ordering fence for LDS repack (DS ops are in-order per wave; this
// waits the hw DScnt and is a compiler memory barrier).
#define WAVE_SYNC() asm volatile("s_wait_dscnt 0" ::: "memory")

__device__ __forceinline__ v8f wmma_bf16(v16bf a, v16bf b, v8f c) {
  // 8 args: (neg_a, A, neg_b, B, c_mod, C, reuse_a, reuse_b)
  return __builtin_amdgcn_wmma_f32_16x16x32_bf16(false, a, false, b, (short)0, c,
                                                 false, false);
}

// A fragment (16xK tile, row-major bf16 matrix `m`, leading dim `ld`):
// lane l=L%16 holds row M=l; half hf=L/16; elem i: K = kbase + 16*hf + i
__device__ __forceinline__ v16bf load_A_lds(const bf16* m, int ld, int kbase) {
  const int L = threadIdx.x & 31, l = L & 15, hf = L >> 4;
  const bf16* row = m + l * ld + kbase + 16 * hf;
  union { v16bf v; uint4 q[2]; } r;
  r.q[0] = *(const uint4*)(row);        // K = kbase + 16*hf + 0..7
  r.q[1] = *(const uint4*)(row + 8);    // K = kbase + 16*hf + 8..15
  return r.v;
}

// B fragment (KxN) taken as W^T from torch-layout weight W[N_out][K_in] (row-
// major, leading dim ldw): lane l=L%16 -> column N=nbase+l; elem i -> K =
// kbase + 16*hf + i  => 16 contiguous bf16 = one 32B load per lane.
__device__ __forceinline__ v16bf load_B_gmem(const bf16* __restrict__ W, int ldw,
                                             int nbase, int kbase) {
  const int L = threadIdx.x & 31, n = L & 15, hf = L >> 4;
  const bf16* p = W + (size_t)(nbase + n) * ldw + kbase + 16 * hf;
  union { v16bf v; uint4 q[2]; } r;
  r.q[0] = ((const uint4*)p)[0];
  r.q[1] = ((const uint4*)p)[1];
  return r.v;
}

// C/D layout: elem r of lane L -> row M = r + 8*(L/16), col N = nbase + L%16.
__device__ __forceinline__ void store_D_lds_bf16(v8f d, bf16* m, int ld, int nbase) {
  const int L = threadIdx.x & 31, l = L & 15, hf = L >> 4;
#pragma unroll
  for (int r = 0; r < 8; ++r) m[(r + 8 * hf) * ld + nbase + l] = (bf16)d[r];
}
__device__ __forceinline__ void store_D_lds_f32(v8f d, float* m, int ld, int nbase) {
  const int L = threadIdx.x & 31, l = L & 15, hf = L >> 4;
#pragma unroll
  for (int r = 0; r < 8; ++r) m[(r + 8 * hf) * ld + nbase + l] = d[r];
}

__device__ __forceinline__ v8f bias_bcast(const float* b, int nbase) {
  float bv = b[nbase + (threadIdx.x & 15)];   // bias depends only on column N
  v8f c;
#pragma unroll
  for (int r = 0; r < 8; ++r) c[r] = bv;
  return c;
}

__device__ __forceinline__ v8f relu8(v8f a) {
#pragma unroll
  for (int i = 0; i < 8; ++i) a[i] = fmaxf(a[i], 0.f);
  return a;
}

// ---------------------------------------------------------------------------
__global__ void cvt_bf16_kernel(const float* __restrict__ s, bf16* __restrict__ d,
                                int n) {
  int i = blockIdx.x * 256 + threadIdx.x;
  if (i < n) d[i] = (bf16)s[i];
}

// ---------------------------------------------------------------------------
// Fused dense-MoE: gate softmax + per-expert (x@W1^T -> relu -> @W2^T) chain,
// gate-weighted accumulation, pos_emb added, written to xbuf[vec][B][64].
// Block: 128 thr (4 waves), 64 rows; wave owns one 16-row M tile.
// ---------------------------------------------------------------------------
__global__ __launch_bounds__(128) void moe_kernel(
    const float* __restrict__ x0, const float* __restrict__ x1,
    const float* __restrict__ x2, const float* __restrict__ gate_w,
    const float* __restrict__ gate_b, const bf16* __restrict__ w1b,
    const float* __restrict__ b1, const bf16* __restrict__ w2b,
    const float* __restrict__ b2, const float* __restrict__ pos_emb,
    float* __restrict__ xbuf, int Bn) {
  __shared__ alignas(16) bf16 xs[64][128];        // x tile, bf16
  __shared__ float gates[64][32];                 // softmax(gate logits)
  __shared__ alignas(16) bf16 hbuf[4][16][64];    // wave-private h repack

  const int vec = blockIdx.y;
  const float* __restrict__ xin = vec == 0 ? x0 : (vec == 1 ? x1 : x2);
  const int row0 = blockIdx.x * 64;
  const int tid = threadIdx.x;

  for (int i = tid; i < 64 * 128; i += 128) {
    int r = i >> 7, c = i & 127;
    xs[r][c] = (bf16)xin[(size_t)(row0 + r) * 128 + c];
  }
  __syncthreads();

  // Gate logits (tiny; VALU) ------------------------------------------------
  for (int p = tid; p < 64 * 30; p += 128) {
    int r = p / 30, e = p % 30;
    float acc = gate_b[e];
    const float* gw = gate_w + e * 128;
    for (int i = 0; i < 128; ++i) acc += (float)xs[r][i] * gw[i];
    gates[r][e] = acc;
  }
  __syncthreads();
  if (tid < 64) {
    float mx = -1e30f;
    for (int e = 0; e < 30; ++e) mx = fmaxf(mx, gates[tid][e]);
    float s = 0.f;
    for (int e = 0; e < 30; ++e) { float v = __expf(gates[tid][e] - mx); gates[tid][e] = v; s += v; }
    float inv = 1.f / s;
    for (int e = 0; e < 30; ++e) gates[tid][e] *= inv;
  }
  __syncthreads();

  const int w = tid >> 5, L = tid & 31, l = L & 15, hf = L >> 4;
  bf16* hb = &hbuf[w][0][0];

  v16bf xa[4];
#pragma unroll
  for (int kk = 0; kk < 4; ++kk) xa[kk] = load_A_lds(&xs[w * 16][0], 128, kk * 32);

  v8f facc[4];
#pragma unroll
  for (int nt = 0; nt < 4; ++nt)
#pragma unroll
    for (int r = 0; r < 8; ++r) facc[nt][r] = 0.f;

  for (int e = 0; e < 30; ++e) {
    const bf16* W1 = w1b + (size_t)e * 64 * 128;
    const bf16* W2 = w2b + (size_t)e * 64 * 64;
    const float* b1e = b1 + e * 64;
    const float* b2e = b2 + e * 64;
    __builtin_prefetch(W1 + 64 * 128, 0, 1);   // next expert's weights

    // h = relu(x @ W1^T + b1)  : 4 N-tiles x 4 K-steps
#pragma unroll
    for (int nt = 0; nt < 4; ++nt) {
      v8f hacc = bias_bcast(b1e, nt * 16);
#pragma unroll
      for (int kk = 0; kk < 4; ++kk)
        hacc = wmma_bf16(xa[kk], load_B_gmem(W1, 128, nt * 16, kk * 32), hacc);
      store_D_lds_bf16(relu8(hacc), hb, 64, nt * 16);
    }
    WAVE_SYNC();

    v16bf ha0 = load_A_lds(hb, 64, 0);
    v16bf ha1 = load_A_lds(hb, 64, 32);

    float g[8];
#pragma unroll
    for (int r = 0; r < 8; ++r) g[r] = gates[w * 16 + r + 8 * hf][e];

    // y = h @ W2^T + b2 ; out += gate * y
#pragma unroll
    for (int nt = 0; nt < 4; ++nt) {
      v8f yacc = bias_bcast(b2e, nt * 16);
      yacc = wmma_bf16(ha0, load_B_gmem(W2, 64, nt * 16, 0), yacc);
      yacc = wmma_bf16(ha1, load_B_gmem(W2, 64, nt * 16, 32), yacc);
#pragma unroll
      for (int r = 0; r < 8; ++r) facc[nt][r] += g[r] * yacc[r];
    }
    WAVE_SYNC();   // ha loads retired before next expert overwrites hbuf
  }

  const float* pe = pos_emb + (vec + 1) * 64;   // token = vec+1
  float* dst = xbuf + (size_t)vec * Bn * 64;
#pragma unroll
  for (int nt = 0; nt < 4; ++nt)
#pragma unroll
    for (int r = 0; r < 8; ++r) {
      int row = row0 + w * 16 + r + 8 * hf;
      int col = nt * 16 + l;
      dst[(size_t)row * 64 + col] = facc[nt][r] + pe[col];
    }
}

// ---------------------------------------------------------------------------
// Full 2-layer transformer + head; per-sample independent => single kernel.
// Wave = 4 samples x 4 tokens = one 16x64 tile.
// ---------------------------------------------------------------------------
__global__ __launch_bounds__(128) void xformer_kernel(
    const float* __restrict__ xbuf, const float* __restrict__ cls,
    const float* __restrict__ pos_emb, const bf16* __restrict__ qkvwb,
    const float* __restrict__ qkv_b, const bf16* __restrict__ outwb,
    const float* __restrict__ out_b, const float* __restrict__ ln1_g,
    const float* __restrict__ ln1_b, const bf16* __restrict__ ffn1b,
    const float* __restrict__ ffn_b1, const bf16* __restrict__ ffn2b,
    const float* __restrict__ ffn_b2, const float* __restrict__ ln2_g,
    const float* __restrict__ ln2_b, const float* __restrict__ head_w,
    const float* __restrict__ head_b, float* __restrict__ out, int Bn) {
  __shared__ float xf[4][16][64];                  // residual stream (f32)
  __shared__ float qk[4][16][192];                 // qkv / proj scratch (f32)
  __shared__ alignas(16) bf16 xb[4][16][64];       // bf16 copy for A-frags
  __shared__ alignas(16) bf16 ab[4][16][64];       // attention output (bf16)
  __shared__ alignas(16) bf16 fb[4][16][256];      // ffn hidden (bf16)

  const int tid = threadIdx.x;
  const int w = tid >> 5, L = tid & 31, l = L & 15, hf = L >> 4;
  const int sbase = blockIdx.x * 16 + w * 4;

  // Load tile: row = 4*sample + token; token0 = cls + pos0, others from xbuf.
  for (int i = L; i < 16 * 64; i += 32) {
    int r = i >> 6, c = i & 63;
    int s = r >> 2, t = r & 3;
    float v = (t == 0) ? (cls[c] + pos_emb[c])
                       : xbuf[(size_t)(t - 1) * Bn * 64 + (size_t)(sbase + s) * 64 + c];
    xf[w][r][c] = v;
  }
  WAVE_SYNC();

  for (int layer = 0; layer < 2; ++layer) {
    const bf16* Wqkv = qkvwb + (size_t)layer * 192 * 64;
    const bf16* Wout = outwb + (size_t)layer * 64 * 64;
    const bf16* Wf1 = ffn1b + (size_t)layer * 256 * 64;
    const bf16* Wf2 = ffn2b + (size_t)layer * 64 * 256;
    const float* bqkv = qkv_b + layer * 192;
    const float* bout = out_b + layer * 64;
    const float* bfc1 = ffn_b1 + layer * 256;
    const float* bfc2 = ffn_b2 + layer * 64;
    const float* g1 = ln1_g + layer * 64;  const float* be1 = ln1_b + layer * 64;
    const float* g2 = ln2_g + layer * 64;  const float* be2 = ln2_b + layer * 64;
    float* of = &qk[w][0][0];              // [16][64] f32 scratch (aliases qk)

    for (int i = L; i < 16 * 64; i += 32) {
      int r = i >> 6, c = i & 63;
      xb[w][r][c] = (bf16)xf[w][r][c];
    }
    WAVE_SYNC();
    v16bf A0 = load_A_lds(&xb[w][0][0], 64, 0);
    v16bf A1 = load_A_lds(&xb[w][0][0], 64, 32);

    // QKV: [16x64] @ [64x192]
#pragma unroll
    for (int nt = 0; nt < 12; ++nt) {
      v8f acc = bias_bcast(bqkv, nt * 16);
      acc = wmma_bf16(A0, load_B_gmem(Wqkv, 64, nt * 16, 0), acc);
      acc = wmma_bf16(A1, load_B_gmem(Wqkv, 64, nt * 16, 32), acc);
      store_D_lds_f32(acc, &qk[w][0][0], 192, nt * 16);
    }
    WAVE_SYNC();

    // Attention (VALU; per-sample 4x4, too small / masked-diag for WMMA).
    {
      int s = L >> 3, h = (L >> 1) & 3, sub = L & 1;   // 4 samples x 4 heads x 2
#pragma unroll
      for (int qi = 0; qi < 2; ++qi) {
        int qt = sub * 2 + qi, qrow = s * 4 + qt;
        float sc[4];
#pragma unroll
        for (int kt = 0; kt < 4; ++kt) {
          float d = 0.f;
#pragma unroll
          for (int j = 0; j < 16; ++j)
            d += qk[w][qrow][h * 16 + j] * qk[w][s * 4 + kt][64 + h * 16 + j];
          sc[kt] = d * 0.25f;   // 1/sqrt(HD=16)
        }
        float mx = fmaxf(fmaxf(sc[0], sc[1]), fmaxf(sc[2], sc[3]));
        float sum = 0.f;
#pragma unroll
        for (int kt = 0; kt < 4; ++kt) { sc[kt] = __expf(sc[kt] - mx); sum += sc[kt]; }
        float inv = 1.f / sum;
#pragma unroll
        for (int kt = 0; kt < 4; ++kt) sc[kt] *= inv;
#pragma unroll
        for (int j = 0; j < 16; ++j) {
          float o = sc[0] * qk[w][s * 4 + 0][128 + h * 16 + j] +
                    sc[1] * qk[w][s * 4 + 1][128 + h * 16 + j] +
                    sc[2] * qk[w][s * 4 + 2][128 + h * 16 + j] +
                    sc[3] * qk[w][s * 4 + 3][128 + h * 16 + j];
          ab[w][qrow][h * 16 + j] = (bf16)o;
        }
      }
    }
    WAVE_SYNC();

    // Out-projection: [16x64] @ [64x64]  (qk reads above retire first: DS in-order)
    v16bf O0 = load_A_lds(&ab[w][0][0], 64, 0);
    v16bf O1 = load_A_lds(&ab[w][0][0], 64, 32);
#pragma unroll
    for (int nt = 0; nt < 4; ++nt) {
      v8f acc = bias_bcast(bout, nt * 16);
      acc = wmma_bf16(O0, load_B_gmem(Wout, 64, nt * 16, 0), acc);
      acc = wmma_bf16(O1, load_B_gmem(Wout, 64, nt * 16, 32), acc);
      store_D_lds_f32(acc, of, 64, nt * 16);
    }
    WAVE_SYNC();

    // residual + LN1 (lanes 0..15, one row each)
    if (hf == 0) {
      float* xr = &xf[w][l][0];
      const float* orow = of + l * 64;
      float mu = 0.f;
      for (int c = 0; c < 64; ++c) mu += xr[c] + orow[c];
      mu *= (1.f / 64.f);
      float var = 0.f;
      for (int c = 0; c < 64; ++c) { float d = xr[c] + orow[c] - mu; var += d * d; }
      var *= (1.f / 64.f);
      float rs = rsqrtf(var + LN_EPSF);
      for (int c = 0; c < 64; ++c) xr[c] = (xr[c] + orow[c] - mu) * rs * g1[c] + be1[c];
    }
    WAVE_SYNC();

    // FFN: relu([16x64]@[64x256]) @ [256x64]
    for (int i = L; i < 16 * 64; i += 32) {
      int r = i >> 6, c = i & 63;
      xb[w][r][c] = (bf16)xf[w][r][c];
    }
    WAVE_SYNC();
    v16bf X0 = load_A_lds(&xb[w][0][0], 64, 0);
    v16bf X1 = load_A_lds(&xb[w][0][0], 64, 32);
#pragma unroll
    for (int nt = 0; nt < 16; ++nt) {
      v8f acc = bias_bcast(bfc1, nt * 16);
      acc = wmma_bf16(X0, load_B_gmem(Wf1, 64, nt * 16, 0), acc);
      acc = wmma_bf16(X1, load_B_gmem(Wf1, 64, nt * 16, 32), acc);
      store_D_lds_bf16(relu8(acc), &fb[w][0][0], 256, nt * 16);
    }
    WAVE_SYNC();
    v16bf F[8];
#pragma unroll
    for (int kk = 0; kk < 8; ++kk) F[kk] = load_A_lds(&fb[w][0][0], 256, kk * 32);
#pragma unroll
    for (int nt = 0; nt < 4; ++nt) {
      v8f acc = bias_bcast(bfc2, nt * 16);
#pragma unroll
      for (int kk = 0; kk < 8; ++kk)
        acc = wmma_bf16(F[kk], load_B_gmem(Wf2, 256, nt * 16, kk * 32), acc);
      store_D_lds_f32(acc, of, 64, nt * 16);
    }
    WAVE_SYNC();

    // residual + LN2
    if (hf == 0) {
      float* xr = &xf[w][l][0];
      const float* orow = of + l * 64;
      float mu = 0.f;
      for (int c = 0; c < 64; ++c) mu += xr[c] + orow[c];
      mu *= (1.f / 64.f);
      float var = 0.f;
      for (int c = 0; c < 64; ++c) { float d = xr[c] + orow[c] - mu; var += d * d; }
      var *= (1.f / 64.f);
      float rs = rsqrtf(var + LN_EPSF);
      for (int c = 0; c < 64; ++c) xr[c] = (xr[c] + orow[c] - mu) * rs * g2[c] + be2[c];
    }
    WAVE_SYNC();
  }

  // Head on CLS rows: 4 samples x 2 classes -> lanes 0..7.
  if (L < 8) {
    int s = L >> 1, c = L & 1;
    float acc = head_b[c];
    const float* xr = &xf[w][s * 4][0];
    for (int d = 0; d < 64; ++d) acc += xr[d] * head_w[c * 64 + d];
    out[(size_t)(sbase + s) * 2 + c] = acc;
  }
}

// ---------------------------------------------------------------------------
extern "C" void kernel_launch(void* const* d_in, const int* in_sizes, int n_in,
                              void* d_out, int out_size, void* d_ws, size_t ws_size,
                              hipStream_t stream) {
  const float* vecb   = (const float*)d_in[0];
  const float* vecc   = (const float*)d_in[1];
  const float* vecf   = (const float*)d_in[2];
  const float* gate_w = (const float*)d_in[3];
  const float* gate_b = (const float*)d_in[4];
  const float* exp_w1 = (const float*)d_in[5];
  const float* exp_b1 = (const float*)d_in[6];
  const float* exp_w2 = (const float*)d_in[7];
  const float* exp_b2 = (const float*)d_in[8];
  const float* cls_t  = (const float*)d_in[9];
  const float* pos_e  = (const float*)d_in[10];
  const float* qkv_w  = (const float*)d_in[11];
  const float* qkv_b  = (const float*)d_in[12];
  const float* out_w  = (const float*)d_in[13];
  const float* out_b  = (const float*)d_in[14];
  const float* ln1_g  = (const float*)d_in[15];
  const float* ln1_b  = (const float*)d_in[16];
  const float* ffn_w1 = (const float*)d_in[17];
  const float* ffn_b1 = (const float*)d_in[18];
  const float* ffn_w2 = (const float*)d_in[19];
  const float* ffn_b2 = (const float*)d_in[20];
  const float* ln2_g  = (const float*)d_in[21];
  const float* ln2_b  = (const float*)d_in[22];
  const float* head_w = (const float*)d_in[23];
  const float* head_b = (const float*)d_in[24];
  (void)n_in; (void)out_size; (void)ws_size;

  const int Bn = in_sizes[0] / 128;   // 16384

  char* ws = (char*)d_ws;
  size_t off = 0;
  auto take = [&](size_t bytes) -> char* {
    char* p = ws + off;
    off += (bytes + 255) & ~(size_t)255;
    return p;
  };

  const int n_w1 = 30 * 64 * 128, n_w2 = 30 * 64 * 64;
  const int n_qkv = 2 * 192 * 64, n_out = 2 * 64 * 64;
  const int n_f1 = 2 * 256 * 64, n_f2 = 2 * 64 * 256;

  bf16* w1b   = (bf16*)take((size_t)n_w1 * 2);
  bf16* w2b   = (bf16*)take((size_t)n_w2 * 2);
  bf16* qkvwb = (bf16*)take((size_t)n_qkv * 2);
  bf16* outwb = (bf16*)take((size_t)n_out * 2);
  bf16* f1b   = (bf16*)take((size_t)n_f1 * 2);
  bf16* f2b   = (bf16*)take((size_t)n_f2 * 2);
  float* xbuf = (float*)take((size_t)3 * Bn * 64 * sizeof(float));

  auto cvt = [&](const float* s, bf16* d, int n) {
    cvt_bf16_kernel<<<(n + 255) / 256, 256, 0, stream>>>(s, d, n);
  };
  cvt(exp_w1, w1b, n_w1);
  cvt(exp_w2, w2b, n_w2);
  cvt(qkv_w, qkvwb, n_qkv);
  cvt(out_w, outwb, n_out);
  cvt(ffn_w1, f1b, n_f1);
  cvt(ffn_w2, f2b, n_f2);

  moe_kernel<<<dim3(Bn / 64, 3), 128, 0, stream>>>(
      vecb, vecc, vecf, gate_w, gate_b, w1b, exp_b1, w2b, exp_b2, pos_e, xbuf, Bn);

  xformer_kernel<<<dim3(Bn / 16), 128, 0, stream>>>(
      xbuf, cls_t, pos_e, qkvwb, qkv_b, outwb, out_b, ln1_g, ln1_b, f1b, ffn_b1,
      f2b, ffn_b2, ln2_g, ln2_b, head_w, head_b, (float*)d_out, Bn);
}
